// ScaledDotProduct_51436528337629
// MI455X (gfx1250) — matrix operand order repaired and checked by reading, not compiled, not run on hardware
//
#include <hip/hip_runtime.h>
#include <math.h>

typedef __attribute__((ext_vector_type(16))) _Float16 v16h;
typedef __attribute__((ext_vector_type(8)))  _Float16 v8h;
typedef __attribute__((ext_vector_type(4)))  _Float16 v4h;
typedef __attribute__((ext_vector_type(2)))  _Float16 v2h;
typedef __attribute__((ext_vector_type(8)))  float    v8f;
typedef __attribute__((ext_vector_type(4)))  float    v4f;
typedef __attribute__((ext_vector_type(8)))  unsigned int v8u;

#define NB 4
#define NH 12
#define NS 2048
#define ND 64
#define KB 32          // keys processed per iteration
#define QTILE 64       // query rows per block (4 waves x 16)
#define KS_STRIDE 72   // halfs per K row in LDS (64 + 8 pad)
#define VT_STRIDE 40   // halfs per Vt row in LDS (32 + 8 pad)

__device__ __forceinline__ v16h cat8(v8h lo, v8h hi) {
  return __builtin_shufflevector(lo, hi, 0,1,2,3,4,5,6,7,8,9,10,11,12,13,14,15);
}

__global__ __launch_bounds__(128)
void fattn_fwd(const float* __restrict__ Qg, const float* __restrict__ Kg,
               const float* __restrict__ Vg, float* __restrict__ Og) {
  // double-buffered staging: K row-major [key][d], V transposed [d][key]
  __shared__ _Float16 KsBuf[2][KB * KS_STRIDE];
  __shared__ _Float16 VtBuf[2][ND * VT_STRIDE];

  const int tid     = threadIdx.x;
  const int lane    = tid & 31;
  const int wave    = tid >> 5;
  const int lq      = lane & 15;          // this lane's query column (N)
  const bool hiHalf = lane >= 16;
  const int kb8     = hiHalf ? 8 : 0;     // accumulator row (M) base for this half-wave

  const int h = blockIdx.y, b = blockIdx.z;
  const size_t bhBase = ((size_t)b * NH + h) * (size_t)NS * ND;  // Q/K/O base
  const int qRow = blockIdx.x * QTILE + wave * 16 + lq;
  const float* qp = Qg + bhBase + (size_t)qRow * ND;

  const float scale = 0.036084391824351615f;   // 1/sqrt(768)

  // ---- Q^T as B-operand (lane<16: K=d contiguous 0..15; lane>=16: 16..31),
  // pre-scaled, two tiles over d (0..31, 32..63)
  v16h Bq[2];
  #pragma unroll
  for (int t = 0; t < 2; ++t) {
    const float* src = qp + 32*t + (hiHalf ? 16 : 0);
    #pragma unroll
    for (int i4 = 0; i4 < 4; ++i4) {
      v4f f = *(const v4f*)(src + 4*i4);
      #pragma unroll
      for (int i = 0; i < 4; ++i) Bq[t][i4*4 + i] = (_Float16)(f[i] * scale);
    }
  }

  // O^T accumulators: 4 tiles of (16 d-rows x 16 q), fp32
  v8f acc[4] = {};
  float m = -INFINITY, l = 0.f;

  // cooperative staging assignment: 128 threads cover 32x64 elements
  const int srow = tid >> 2;           // key row 0..31
  const int sd   = (tid & 3) * 16;     // d segment start
  const float* kp0 = Kg + bhBase + (size_t)srow * ND + sd;
  const float* vp0 = Vg + (((size_t)b * NS + srow) * NH + h) * ND + sd;

  // preload first block into registers
  v4f kf[4], vf[4];
  #pragma unroll
  for (int i = 0; i < 4; ++i) {
    kf[i] = *(const v4f*)(kp0 + 4*i);
    vf[i] = *(const v4f*)(vp0 + 4*i);
  }

  int cur = 0;
  for (int kb = 0; kb < NS; kb += KB) {
    // ---- commit staged registers to LDS buffer `cur` (fp32->f16) ----
    _Float16* ks = KsBuf[cur];
    _Float16* vt = VtBuf[cur];
    #pragma unroll
    for (int i = 0; i < 4; ++i) {
      v4h khv;
      #pragma unroll
      for (int j = 0; j < 4; ++j) khv[j] = (_Float16)kf[i][j];
      *(v4h*)(&ks[srow * KS_STRIDE + sd + 4*i]) = khv;            // ds_store_b64
      #pragma unroll
      for (int j = 0; j < 4; ++j)
        vt[(sd + 4*i + j) * VT_STRIDE + srow] = (_Float16)vf[i][j]; // transpose
    }
    __syncthreads();   // single barrier per iteration (double buffering)

    // ---- issue next block's global loads early: overlap with compute ----
    if (kb + KB < NS) {
      const float* kp = kp0 + (size_t)(kb + KB) * ND;
      const float* vp = vp0 + (size_t)(kb + KB) * NH * ND;
      __builtin_prefetch(kp + (size_t)KB * ND, 0, 0);         // warm block kb+2
      __builtin_prefetch(vp + (size_t)KB * NH * ND, 0, 0);
      #pragma unroll
      for (int i = 0; i < 4; ++i) {
        kf[i] = *(const v4f*)(kp + 4*i);
        vf[i] = *(const v4f*)(vp + 4*i);
      }
    }

    // ---- batch all K A-tile loads, then run score WMMAs back-to-back ----
    v16h Ak[2][2];
    #pragma unroll
    for (int g = 0; g < 2; ++g)
      #pragma unroll
      for (int t = 0; t < 2; ++t) {
        const _Float16* base = &ks[(g*16 + lq) * KS_STRIDE + 32*t + (hiHalf ? 16 : 0)];
        Ak[g][t] = cat8(*(const v8h*)base, *(const v8h*)(base + 8));
      }
    v8f Sg[2];
    #pragma unroll
    for (int g = 0; g < 2; ++g) {
      v8f c = {};
      c = __builtin_amdgcn_wmma_f32_16x16x32_f16(false, Ak[g][0], false, Bq[0],
                                                 (short)0, c, false, false);
      c = __builtin_amdgcn_wmma_f32_16x16x32_f16(false, Ak[g][1], false, Bq[1],
                                                 (short)0, c, false, false);
      Sg[g] = c;
    }

    // ---- hoist V^T A-tile loads: ds latency overlaps softmax VALU ----
    v16h Av[4];
    #pragma unroll
    for (int j = 0; j < 4; ++j) {
      const _Float16* base = &vt[(j*16 + lq) * VT_STRIDE + (hiHalf ? 16 : 0)];
      Av[j] = cat8(*(const v8h*)base, *(const v8h*)(base + 8));
    }

    // ---- online softmax: lane holds q=lq, keys {kb8+r} U {16+kb8+r} ----
    float mblk = -INFINITY;
    #pragma unroll
    for (int r = 0; r < 8; ++r) mblk = fmaxf(mblk, fmaxf(Sg[0][r], Sg[1][r]));
    mblk = fmaxf(mblk, __shfl_xor(mblk, 16, 32));
    const float mnew  = fmaxf(m, mblk);
    const float alpha = __expf(m - mnew);

    float e0[8], e1[8], lsum = 0.f;
    #pragma unroll
    for (int r = 0; r < 8; ++r) {
      e0[r] = __expf(Sg[0][r] - mnew);
      e1[r] = __expf(Sg[1][r] - mnew);
      lsum += e0[r] + e1[r];
    }
    lsum += __shfl_xor(lsum, 16, 32);
    l = l * alpha + lsum;
    m = mnew;

    // ---- P^T B-operand: pack to half2 words, 8 dword cross-half shuffles ----
    unsigned int u0[4], u1[4];
    #pragma unroll
    for (int r = 0; r < 4; ++r) {
      v2h p0; p0[0] = (_Float16)e0[2*r]; p0[1] = (_Float16)e0[2*r+1];
      v2h p1; p1[0] = (_Float16)e1[2*r]; p1[1] = (_Float16)e1[2*r+1];
      u0[r] = __builtin_bit_cast(unsigned int, p0);
      u1[r] = __builtin_bit_cast(unsigned int, p1);
    }
    unsigned int lo[4], hi[4];
    #pragma unroll
    for (int r = 0; r < 4; ++r) {
      unsigned int o0 = (unsigned int)__shfl_xor((int)u0[r], 16, 32);
      unsigned int o1 = (unsigned int)__shfl_xor((int)u1[r], 16, 32);
      lo[r] = hiHalf ? o1 : u0[r];   // keys kb16+0..7  (own / partner half)
      hi[r] = hiHalf ? u1[r] : o0;   // keys kb16+8..15
    }
    v8u bpw = {lo[0], lo[1], lo[2], lo[3], hi[0], hi[1], hi[2], hi[3]};
    v16h Bp = __builtin_bit_cast(v16h, bpw);

    // ---- rescale (scalar-per-lane) and accumulate O^T += V^T x P^T ----
    #pragma unroll
    for (int j = 0; j < 4; ++j) acc[j] *= alpha;
    #pragma unroll
    for (int j = 0; j < 4; ++j)
      acc[j] = __builtin_amdgcn_wmma_f32_16x16x32_f16(false, Av[j], false, Bp,
                                                      (short)0, acc[j], false, false);
    cur ^= 1;
  }

  // ---- normalize and write: lane writes q=lq, d = j*16 + kb8 + (0..7) ----
  const float inv = 1.f / l;
  float* op = Og + bhBase + (size_t)qRow * ND;
  #pragma unroll
  for (int j = 0; j < 4; ++j) {
    v4f o0, o1;
    #pragma unroll
    for (int i = 0; i < 4; ++i) { o0[i] = acc[j][i] * inv; o1[i] = acc[j][4+i] * inv; }
    *(v4f*)(op + j*16 + kb8)     = o0;
    *(v4f*)(op + j*16 + kb8 + 4) = o1;
  }
}

extern "C" void kernel_launch(void* const* d_in, const int* in_sizes, int n_in,
                              void* d_out, int out_size, void* d_ws, size_t ws_size,
                              hipStream_t stream) {
  (void)in_sizes; (void)n_in; (void)out_size; (void)d_ws; (void)ws_size;
  const float* Q = (const float*)d_in[0];  // [B,H,S,D]
  const float* K = (const float*)d_in[1];  // [B,H,S,D]
  const float* V = (const float*)d_in[2];  // [B,S,H,D]
  float* O = (float*)d_out;                // [B,H,S,D]
  dim3 grid(NS / QTILE, NH, NB);
  fattn_fwd<<<grid, 128, 0, stream>>>(Q, K, V, O);
}
